// GraphNet_82454782149339
// MI455X (gfx1250) — matrix operand order repaired and checked by reading, not compiled, not run on hardware
//
#include <hip/hip_runtime.h>
#include <hip/hip_bf16.h>
#include <math.h>

// ---------------------------------------------------------------------------
// Types for CDNA5 WMMA (gfx1250, wave32)
// ---------------------------------------------------------------------------
typedef __attribute__((ext_vector_type(16))) __bf16 v16bf;
typedef __attribute__((ext_vector_type(8)))  __bf16 v8bf;
typedef __attribute__((ext_vector_type(8)))  float  v8f;

#define EPS_BN 1e-5f

// ---------------------------------------------------------------------------
// Small device helpers
// ---------------------------------------------------------------------------
__device__ __forceinline__ float eluf(float x)   { return x > 0.f ? x : (__expf(x) - 1.f); }
__device__ __forceinline__ float leakyf(float x, float s) { return x > 0.f ? x : s * x; }

__device__ __forceinline__ void atomicMaxF(float* addr, float val) {
    unsigned* ua = (unsigned*)addr;
    unsigned old = *ua;
    while (__uint_as_float(old) < val) {
        unsigned assumed = old;
        old = atomicCAS(ua, assumed, __float_as_uint(val));
        if (old == assumed) break;
    }
}

__device__ __forceinline__ void atomicMulF(float* addr, float val) {
    unsigned* ua = (unsigned*)addr;
    unsigned old = *ua, assumed;
    do {
        assumed = old;
        float nv = __uint_as_float(assumed) * val;
        old = atomicCAS(ua, assumed, __float_as_uint(nv));
    } while (old != assumed);
}

// edge e in [0, E) -> real edge from edge_index; e in [E, E+N) -> self loop
__device__ __forceinline__ void edge_sd(const int* __restrict__ ei, int e, int E,
                                        int& s, int& d) {
    if (e < E) { s = ei[e]; d = ei[E + e]; }
    else       { s = d = e - E; }
}

// ---------------------------------------------------------------------------
// Fill
// ---------------------------------------------------------------------------
__global__ void k_fill(float* __restrict__ p, float v, int n) {
    int i = blockIdx.x * blockDim.x + threadIdx.x;
    if (i < n) p[i] = v;
}

// ---------------------------------------------------------------------------
// Weight -> bf16 WMMA B-fragment layout (with K zero-padding).
// Fragment blocks of 512 bf16: block = (k0/32)*(Nc/16) + tile16
//   element (lane, idx): W[k0 + (idx<8 ? 8*(lane>>4) + idx : 16 + 8*(lane>>4) + idx - 8)][tile16*16 + (lane&15)]
// ---------------------------------------------------------------------------
__global__ void k_w_frag(const float* __restrict__ W, __bf16* __restrict__ Wf,
                         int Ksrc, int Kpad, int Nc) {
    int i = blockIdx.x * blockDim.x + threadIdx.x;
    int total = Kpad * Nc;
    if (i >= total) return;
    int block = i >> 9, r = i & 511;
    int lane = r >> 4, idx = r & 15;
    int nt16 = Nc >> 4;
    int k0  = (block / nt16) << 5;
    int t16 = block % nt16;
    int k   = k0 + ((idx & 8) << 1) + ((lane >> 4) << 3) + (idx & 7);
    int col = (t16 << 4) + (lane & 15);
    Wf[i] = (k < Ksrc) ? (__bf16)W[(size_t)k * Nc + col] : (__bf16)0.f;
}

// pad columns into bf16: dst[M,Kp] from src[M,K] (f32)
__global__ void k_pad_cols_bf16(const float* __restrict__ src, __bf16* __restrict__ dst,
                                int M, int K, int Kp) {
    int i = blockIdx.x * blockDim.x + threadIdx.x;
    if (i >= M * Kp) return;
    int m = i / Kp, j = i % Kp;
    dst[i] = (j < K) ? (__bf16)src[m * K + j] : (__bf16)0.f;
}

// ---------------------------------------------------------------------------
// WMMA GEMM, all-bf16 operands, zero in-loop conversions:
//   D[M,Nc] = act(A[M,K] @ W[K,Nc] + bias)
// A: bf16 row-major (K%32==0).  Wf: bf16 fragment layout from k_w_frag.
// M%16==0, Nc%64==0. One wave = 16x64 strip (A fragment reused 4x).
// act: 0=none, 1=elu.  out_bf16: store bf16 else f32.
// ---------------------------------------------------------------------------
__global__ __launch_bounds__(256)
void k_gemm_wmma(const __bf16* __restrict__ A, const __bf16* __restrict__ Wf,
                 const float* __restrict__ bias, void* __restrict__ Dout,
                 int M, int K, int Nc, int act, int out_bf16)
{
    int tilesN = Nc >> 6;                           // 64-wide N strips
    int wave = blockIdx.x * (blockDim.x >> 5) + (threadIdx.x >> 5);
    if (wave >= (M >> 4) * tilesN) return;          // wave-uniform exit
    int tm = wave / tilesN, tn = wave % tilesN;
    int lane = threadIdx.x & 31;
    int l16 = lane & 15, half = lane >> 4;

    v8f acc0, acc1, acc2, acc3;
    #pragma unroll
    for (int i = 0; i < 8; ++i) { acc0[i] = 0.f; acc1[i] = 0.f; acc2[i] = 0.f; acc3[i] = 0.f; }

    int row = tm * 16 + l16;
    const __bf16* Arow = A + (size_t)row * K + 8 * half;
    int nt16 = Nc >> 4;
    const __bf16* Wb = Wf + (size_t)(tn * 4) * 512 + (size_t)lane * 16;
    size_t wstep = (size_t)nt16 * 512;              // fragment stride per k-block

    for (int k0 = 0; k0 < K; k0 += 32) {
        v8bf alo = *(const v8bf*)(Arow + k0);        // K = k0+8h .. +7   (16B)
        v8bf ahi = *(const v8bf*)(Arow + k0 + 16);   // K = k0+16+8h .. +7 (16B)
        v16bf a = __builtin_shufflevector(alo, ahi, 0, 1, 2, 3, 4, 5, 6, 7,
                                          8, 9, 10, 11, 12, 13, 14, 15);
        const __bf16* wb = Wb + (size_t)(k0 >> 5) * wstep;
        v16bf b0 = *(const v16bf*)(wb);
        v16bf b1 = *(const v16bf*)(wb + 512);
        v16bf b2 = *(const v16bf*)(wb + 1024);
        v16bf b3 = *(const v16bf*)(wb + 1536);
        acc0 = __builtin_amdgcn_wmma_f32_16x16x32_bf16(false, a, false, b0, (short)0, acc0, false, false);
        acc1 = __builtin_amdgcn_wmma_f32_16x16x32_bf16(false, a, false, b1, (short)0, acc1, false, false);
        acc2 = __builtin_amdgcn_wmma_f32_16x16x32_bf16(false, a, false, b2, (short)0, acc2, false, false);
        acc3 = __builtin_amdgcn_wmma_f32_16x16x32_bf16(false, a, false, b3, (short)0, acc3, false, false);
    }

    int col0 = tn * 64 + l16;
    float bb0 = bias ? bias[col0]      : 0.f;
    float bb1 = bias ? bias[col0 + 16] : 0.f;
    float bb2 = bias ? bias[col0 + 32] : 0.f;
    float bb3 = bias ? bias[col0 + 48] : 0.f;
    #pragma unroll
    for (int v = 0; v < 8; ++v) {
        int r = tm * 16 + v + 8 * half;             // D row mapping (VGPR v, half)
        float x0 = acc0[v] + bb0, x1 = acc1[v] + bb1;
        float x2 = acc2[v] + bb2, x3 = acc3[v] + bb3;
        if (act == 1) { x0 = eluf(x0); x1 = eluf(x1); x2 = eluf(x2); x3 = eluf(x3); }
        size_t base = (size_t)r * Nc + col0;
        if (out_bf16) {
            __bf16* D = (__bf16*)Dout;
            D[base] = (__bf16)x0; D[base + 16] = (__bf16)x1;
            D[base + 32] = (__bf16)x2; D[base + 48] = (__bf16)x3;
        } else {
            float* D = (float*)Dout;
            D[base] = x0; D[base + 16] = x1; D[base + 32] = x2; D[base + 48] = x3;
        }
    }
}

// ---------------------------------------------------------------------------
// Preprocessing
// ---------------------------------------------------------------------------
__global__ void k_loop_attr(const float* __restrict__ ea, const int* __restrict__ ei,
                            float* __restrict__ loop, int E) {
    int e = blockIdx.x * blockDim.x + threadIdx.x;
    if (e < E) atomicAdd(&loop[ei[E + e]], ea[e]);
}

__global__ void k_pre_edge(const float* __restrict__ edge_attr,
                           const float* __restrict__ loop,
                           const float* __restrict__ w, const float* __restrict__ b,
                           const float* __restrict__ bnw, const float* __restrict__ bnb,
                           float* __restrict__ ea2, int E, int EN) {
    int i = blockIdx.x * blockDim.x + threadIdx.x;
    if (i >= EN * 2) return;
    int e = i >> 1, j = i & 1;
    float raw = (e < E) ? edge_attr[e] : loop[e - E];
    float v = eluf(raw * w[j] + b[j]);
    ea2[i] = v * rsqrtf(1.f + EPS_BN) * bnw[j] + bnb[j];
}

// xn stored bf16, K-padded to 32 columns (cols 16..31 = 0)
__global__ void k_pre_node(const float* __restrict__ x,
                           const float* __restrict__ w, const float* __restrict__ b,
                           const float* __restrict__ bnw, const float* __restrict__ bnb,
                           __bf16* __restrict__ xn, int N) {
    int i = blockIdx.x * blockDim.x + threadIdx.x;
    if (i >= N * 32) return;
    int n = i >> 5, j = i & 31;
    float v = 0.f;
    if (j < 16) {
        v = eluf(x[n] * w[j] + b[j]);
        v = v * rsqrtf(1.f + EPS_BN) * bnw[j] + bnb[j];
    }
    xn[i] = (__bf16)v;
}

// ---------------------------------------------------------------------------
// GATv2 attention: one block per edge, 8 waves = 8 heads, lane-reduce over C=128
// ---------------------------------------------------------------------------
__global__ __launch_bounds__(256)
void k_gat_score(const __bf16* __restrict__ xl, const __bf16* __restrict__ xr,
                 const int* __restrict__ ei, int E, int EN,
                 const float* __restrict__ att,     // [8*128]
                 const float* __restrict__ we,      // [8*1024] or null
                 const float* __restrict__ ap,      // [EN*8]  or null
                 float* __restrict__ score,         // [EN*8]
                 float* __restrict__ smax)          // [N*8]
{
    int e = blockIdx.x;
    if (e >= EN) return;
    int h = threadIdx.x >> 5, lane = threadIdx.x & 31;
    int s, d; edge_sd(ei, e, E, s, d);
    const __bf16* xls = xl + (size_t)s * 1024 + h * 128;
    const __bf16* xrd = xr + (size_t)d * 1024 + h * 128;
    float apv[8];
    if (we) {
        #pragma unroll
        for (int j = 0; j < 8; ++j) apv[j] = ap[(size_t)e * 8 + j];
    }
    float acc = 0.f;
    #pragma unroll
    for (int cc = 0; cc < 4; ++cc) {
        int c = lane + cc * 32;
        float m = (float)xls[c] + (float)xrd[c];
        if (we) {
            float t = 0.f;
            #pragma unroll
            for (int j = 0; j < 8; ++j) t += apv[j] * we[j * 1024 + h * 128 + c];
            m += t;
        }
        m = leakyf(m, 0.2f);
        acc += m * att[h * 128 + c];
    }
    #pragma unroll
    for (int off = 16; off > 0; off >>= 1) acc += __shfl_xor(acc, off, 32);
    if (lane == 0) {
        score[(size_t)e * 8 + h] = acc;
        atomicMaxF(&smax[(size_t)d * 8 + h], acc);
    }
}

__global__ void k_softmax_exp(float* __restrict__ score, const float* __restrict__ smax,
                              float* __restrict__ ssum, const int* __restrict__ ei,
                              int E, int EN) {
    int i = blockIdx.x * blockDim.x + threadIdx.x;
    if (i >= EN * 8) return;
    int e = i >> 3, h = i & 7;
    int d = (e < E) ? ei[E + e] : (e - E);
    float ex = __expf(score[i] - smax[(size_t)d * 8 + h]);
    score[i] = ex;
    atomicAdd(&ssum[(size_t)d * 8 + h], ex);
}

__global__ void k_softmax_norm(float* __restrict__ score, const float* __restrict__ ssum,
                               const int* __restrict__ ei, int E, int EN) {
    int i = blockIdx.x * blockDim.x + threadIdx.x;
    if (i >= EN * 8) return;
    int e = i >> 3, h = i & 7;
    int d = (e < E) ? ei[E + e] : (e - E);
    score[i] = score[i] / ssum[(size_t)d * 8 + h];
}

__global__ __launch_bounds__(256)
void k_aggregate(const __bf16* __restrict__ xl, const float* __restrict__ alpha,
                 const int* __restrict__ ei, int E, int EN, float* __restrict__ acc) {
    int e = blockIdx.x;
    if (e >= EN) return;
    int s, d; edge_sd(ei, e, E, s, d);
    #pragma unroll
    for (int cc = 0; cc < 4; ++cc) {
        int idx = threadIdx.x + cc * 256;
        int h = idx >> 7;
        atomicAdd(&acc[(size_t)d * 1024 + idx],
                  alpha[(size_t)e * 8 + h] * (float)xl[(size_t)s * 1024 + idx]);
    }
}

__global__ void k_head_mean(const float* __restrict__ acc, const float* __restrict__ bias,
                            float* __restrict__ out, int N) {
    int i = blockIdx.x * blockDim.x + threadIdx.x;
    if (i >= N * 128) return;
    int n = i >> 7, c = i & 127;
    float s = 0.f;
    #pragma unroll
    for (int h = 0; h < 8; ++h) s += acc[(size_t)n * 1024 + h * 128 + c];
    out[i] = s * 0.125f + bias[c];
}

// SimpleConv(aggr=Mul): p[d,c] *= h[s,c]   (p pre-filled with 1.0)
__global__ void k_scatter_mul(const float* __restrict__ h, const int* __restrict__ ei,
                              int E, int EN, float* __restrict__ p) {
    int i = blockIdx.x * blockDim.x + threadIdx.x;
    if (i >= EN * 128) return;
    int e = i >> 7, c = i & 127;
    int s, d; edge_sd(ei, e, E, s, d);
    atomicMulF(&p[(size_t)d * 128 + c], h[(size_t)s * 128 + c]);
}

// leaky + convert to bf16 row-major [N,128] (layer-2 GEMM A operand)
__global__ void k_leaky_to_bf16(const float* __restrict__ src, __bf16* __restrict__ dst,
                                int n, float slope) {
    int i = blockIdx.x * blockDim.x + threadIdx.x;
    if (i < n) dst[i] = (__bf16)leakyf(src[i], slope);
}

// SimpleConv(aggr=[Sum,Mul])
__global__ void k_scatter_sum_mul(const float* __restrict__ h, const int* __restrict__ ei,
                                  int E, int EN, float* __restrict__ ms,
                                  float* __restrict__ mm) {
    int i = blockIdx.x * blockDim.x + threadIdx.x;
    if (i >= EN * 128) return;
    int e = i >> 7, c = i & 127;
    int s, d; edge_sd(ei, e, E, s, d);
    float v = h[(size_t)s * 128 + c];
    atomicAdd(&ms[(size_t)d * 128 + c], v);
    atomicMulF(&mm[(size_t)d * 128 + c], v);
}

__global__ void k_concat_leaky(const float* __restrict__ ms, const float* __restrict__ mm,
                               float* __restrict__ feat, int N) {
    int i = blockIdx.x * blockDim.x + threadIdx.x;
    if (i >= N * 256) return;
    int n = i >> 8, c = i & 255;
    float v = (c < 128) ? ms[(size_t)n * 128 + c] : mm[(size_t)n * 128 + (c - 128)];
    feat[i] = leakyf(v, 0.01f);
}

// GraphNorm(256): one block (256 threads) per graph, one feature per thread
__global__ __launch_bounds__(256)
void k_graphnorm(float* __restrict__ feat, const float* __restrict__ gw,
                 const float* __restrict__ gb, const float* __restrict__ gms, int npg) {
    int g = blockIdx.x, c = threadIdx.x;
    size_t base = (size_t)g * npg * 256;
    float s = 0.f;
    for (int i = 0; i < npg; ++i) s += feat[base + (size_t)i * 256 + c];
    float mm = (s / npg) * gms[c];
    float var = 0.f;
    for (int i = 0; i < npg; ++i) {
        float ce = feat[base + (size_t)i * 256 + c] - mm;
        var += ce * ce;
    }
    float inv = rsqrtf(var / npg + EPS_BN);
    for (int i = 0; i < npg; ++i) {
        float ce = feat[base + (size_t)i * 256 + c] - mm;
        feat[base + (size_t)i * 256 + c] = ce * inv * gw[c] + gb[c];
    }
}

// Edge attention embedding: [ea2(2), alpha2(8)] -> 128 (elu) -> 1
__global__ __launch_bounds__(128)
void k_edge_embed(const float* __restrict__ ea2, const float* __restrict__ alpha2,
                  const float* __restrict__ W1, const float* __restrict__ b1,
                  const float* __restrict__ W2, const float* __restrict__ b2,
                  float* __restrict__ et, int EN) {
    __shared__ float red[128];
    int e = blockIdx.x;
    if (e >= EN) return;
    int c = threadIdx.x;
    float h = b1[c] + ea2[(size_t)e * 2] * W1[c] + ea2[(size_t)e * 2 + 1] * W1[128 + c];
    #pragma unroll
    for (int j = 0; j < 8; ++j) h += alpha2[(size_t)e * 8 + j] * W1[(2 + j) * 128 + c];
    red[c] = eluf(h) * W2[c];
    __syncthreads();
    for (int st = 64; st > 0; st >>= 1) {
        if (c < st) red[c] += red[c + st];
        __syncthreads();
    }
    if (c == 0) et[e] = red[0] + b2[0];
}

// Node pooling sum+mul over graphs, BN eval -> pn[B,512]
__global__ __launch_bounds__(256)
void k_node_pool(const float* __restrict__ feat, const float* __restrict__ w,
                 const float* __restrict__ b, float* __restrict__ pn, int npg) {
    int g = blockIdx.x, c = threadIdx.x;
    size_t base = (size_t)g * npg * 256;
    float s = 0.f, p = 1.f;
    for (int i = 0; i < npg; ++i) {
        float v = feat[base + (size_t)i * 256 + c];
        s += v; p *= v;
    }
    float inv = rsqrtf(1.f + EPS_BN);
    pn[g * 512 + c]       = s * inv * w[c]       + b[c];
    pn[g * 512 + 256 + c] = p * inv * w[256 + c] + b[256 + c];
}

__global__ void k_csum(const float* __restrict__ y, int* __restrict__ csum,
                       int Bg, int GF, int npg) {
    if (blockIdx.x == 0 && threadIdx.x == 0) {
        int c = 0;
        for (int g = 0; g < Bg; ++g) { c += (int)y[g * GF + GF - 1] + npg; csum[g] = c; }
    }
}

__global__ __launch_bounds__(256)
void k_edge_pool(const float* __restrict__ et, const int* __restrict__ csum,
                 const float* __restrict__ w, const float* __restrict__ b,
                 float* __restrict__ pe, int EN) {
    __shared__ float red[256];
    int g = blockIdx.x;
    int lo = g ? csum[g - 1] : 0;
    int hi = csum[g];
    if (lo > EN) lo = EN;
    if (hi > EN) hi = EN;
    float s = 0.f;
    for (int i = lo + (int)threadIdx.x; i < hi; i += 256) s += et[i];
    red[threadIdx.x] = s;
    __syncthreads();
    for (int st = 128; st > 0; st >>= 1) {
        if ((int)threadIdx.x < st) red[threadIdx.x] += red[threadIdx.x + st];
        __syncthreads();
    }
    if (threadIdx.x == 0) pe[g] = red[0] * rsqrtf(1.f + EPS_BN) * w[0] + b[0];
}

// hfeat built bf16, K-padded to 800 columns (769..799 = 0)
__global__ void k_build_h(const float* __restrict__ pn, const float* __restrict__ pe,
                          const float* __restrict__ g, __bf16* __restrict__ hf, int Bg) {
    int i = blockIdx.x * blockDim.x + threadIdx.x;
    if (i >= Bg * 800) return;
    int b = i / 800, j = i % 800;
    float v = 0.f;
    if (j < 512)       v = pn[b * 512 + j];
    else if (j == 512) v = pe[b];
    else if (j < 769)  v = g[b * 256 + (j - 513)];
    hf[i] = (__bf16)v;
}

__global__ void k_f3_sigmoid(const float* __restrict__ h, const float* __restrict__ w,
                             const float* __restrict__ b, float* __restrict__ out, int Bg) {
    int g = blockIdx.x * blockDim.x + threadIdx.x;
    if (g >= Bg) return;
    float s = b[0];
    for (int k = 0; k < 256; ++k) s += h[g * 256 + k] * w[k];
    out[g] = 1.f / (1.f + __expf(-s));
}

// ---------------------------------------------------------------------------
// Host launch
// ---------------------------------------------------------------------------
static inline int cdiv(long long a, long long b) { return (int)((a + b - 1) / b); }

extern "C" void kernel_launch(void* const* d_in, const int* in_sizes, int n_in,
                              void* d_out, int out_size, void* d_ws, size_t ws_size,
                              hipStream_t stream) {
    const int GF = 5;
    const float* x_in   = (const float*)d_in[0];
    const float* ea_in  = (const float*)d_in[1];
    const float* y_in   = (const float*)d_in[2];
    const int*   ei     = (const int*)d_in[3];
    // d_in[4] = batch (contiguous blocks; npg derived below)
    const int N  = in_sizes[0];
    const int E  = in_sizes[1];
    const int Bg = in_sizes[2] / GF;
    const int EN = E + N;
    const int npg = N / Bg;

    // params (insertion order of make_params)
    int P = 5;
    const float* pn1_w = (const float*)d_in[P + 0];
    const float* pn1_b = (const float*)d_in[P + 1];
    const float* bn_n_w = (const float*)d_in[P + 2];
    const float* bn_n_b = (const float*)d_in[P + 3];
    const float* pe1_w = (const float*)d_in[P + 4];
    const float* pe1_b = (const float*)d_in[P + 5];
    const float* bn_e_w = (const float*)d_in[P + 6];
    const float* bn_e_b = (const float*)d_in[P + 7];
    const float* pg1_w = (const float*)d_in[P + 8];
    const float* pg1_b = (const float*)d_in[P + 9];
    const float* pg2_w = (const float*)d_in[P + 10];
    const float* pg2_b = (const float*)d_in[P + 11];
    const float* pg3_w = (const float*)d_in[P + 12];
    const float* pg3_b = (const float*)d_in[P + 13];
    const float* pg4_w = (const float*)d_in[P + 14];
    const float* pg4_b = (const float*)d_in[P + 15];
    const float* g1_l_w = (const float*)d_in[P + 16];
    const float* g1_l_b = (const float*)d_in[P + 17];
    const float* g1_r_w = (const float*)d_in[P + 18];
    const float* g1_r_b = (const float*)d_in[P + 19];
    const float* g1_att = (const float*)d_in[P + 20];
    const float* g1_bias = (const float*)d_in[P + 21];
    const float* g2_l_w = (const float*)d_in[P + 22];
    const float* g2_l_b = (const float*)d_in[P + 23];
    const float* g2_r_w = (const float*)d_in[P + 24];
    const float* g2_r_b = (const float*)d_in[P + 25];
    const float* g2_we = (const float*)d_in[P + 26];
    const float* g2_att = (const float*)d_in[P + 27];
    const float* g2_bias = (const float*)d_in[P + 28];
    const float* gn_w = (const float*)d_in[P + 29];
    const float* gn_b = (const float*)d_in[P + 30];
    const float* gn_ms = (const float*)d_in[P + 31];
    const float* ee1_w = (const float*)d_in[P + 32];
    const float* ee1_b = (const float*)d_in[P + 33];
    const float* ee2_w = (const float*)d_in[P + 34];
    const float* ee2_b = (const float*)d_in[P + 35];
    const float* pnn_w = (const float*)d_in[P + 36];
    const float* pnn_b = (const float*)d_in[P + 37];
    const float* pen_w = (const float*)d_in[P + 38];
    const float* pen_b = (const float*)d_in[P + 39];
    const float* f1_w = (const float*)d_in[P + 40];
    const float* f1_b = (const float*)d_in[P + 41];
    const float* f2_w = (const float*)d_in[P + 42];
    const float* f2_b = (const float*)d_in[P + 43];
    const float* f3_w = (const float*)d_in[P + 44];
    const float* f3_b = (const float*)d_in[P + 45];

    // workspace bump allocator
    char* ws = (char*)d_ws;
    size_t off = 0;
    auto alloc = [&](size_t bytes) -> char* {
        char* p = ws + off;
        off += (bytes + 255) & ~(size_t)255;
        return p;
    };
    float*  loopAttr = (float*)alloc((size_t)N * 4);
    float*  ea2      = (float*)alloc((size_t)EN * 2 * 4);
    __bf16* xn       = (__bf16*)alloc((size_t)N * 32 * 2);         // K-padded 16->32
    __bf16* ygpad    = (__bf16*)alloc((size_t)Bg * 32 * 2);        // K-padded 5->32
    __bf16* gbufA    = (__bf16*)alloc((size_t)Bg * 2048 * 2);
    __bf16* gbufB    = (__bf16*)alloc((size_t)Bg * 2048 * 2);
    float*  gfin     = (float*)alloc((size_t)Bg * 256 * 4);
    __bf16* xl       = (__bf16*)alloc((size_t)N * 1024 * 2);
    __bf16* xr       = (__bf16*)alloc((size_t)N * 1024 * 2);
    float*  acc      = (float*)alloc((size_t)N * 1024 * 4);
    float*  alpha1   = (float*)alloc((size_t)EN * 8 * 4);
    float*  alpha2   = (float*)alloc((size_t)EN * 8 * 4);
    float*  smax     = (float*)alloc((size_t)N * 8 * 4);
    float*  ssum     = (float*)alloc((size_t)N * 8 * 4);
    float*  h1       = (float*)alloc((size_t)N * 128 * 4);
    float*  pbuf     = (float*)alloc((size_t)N * 128 * 4);
    __bf16* h2bf     = (__bf16*)alloc((size_t)N * 128 * 2);
    float*  ho       = (float*)alloc((size_t)N * 128 * 4);
    float*  msum     = (float*)alloc((size_t)N * 128 * 4);
    float*  mmul     = (float*)alloc((size_t)N * 128 * 4);
    float*  feat     = (float*)alloc((size_t)N * 256 * 4);
    float*  et       = (float*)alloc((size_t)EN * 4);
    float*  pn       = (float*)alloc((size_t)Bg * 512 * 4);
    float*  pe       = (float*)alloc((size_t)Bg * 4);
    int*    csum     = (int*)alloc((size_t)Bg * 4);
    __bf16* hfeat    = (__bf16*)alloc((size_t)Bg * 800 * 2);       // K-padded 769->800
    __bf16* f1o      = (__bf16*)alloc((size_t)Bg * 2048 * 2);
    float*  f2o      = (float*)alloc((size_t)Bg * 256 * 4);
    // bf16 fragment-layout weights (K padded to multiple of 32)
    __bf16* wf_pg1 = (__bf16*)alloc((size_t)32 * 256 * 2);
    __bf16* wf_pg2 = (__bf16*)alloc((size_t)256 * 2048 * 2);
    __bf16* wf_pg3 = (__bf16*)alloc((size_t)2048 * 2048 * 2);
    __bf16* wf_pg4 = (__bf16*)alloc((size_t)2048 * 256 * 2);
    __bf16* wf_g1l = (__bf16*)alloc((size_t)32 * 1024 * 2);
    __bf16* wf_g1r = (__bf16*)alloc((size_t)32 * 1024 * 2);
    __bf16* wf_g2l = (__bf16*)alloc((size_t)128 * 1024 * 2);
    __bf16* wf_g2r = (__bf16*)alloc((size_t)128 * 1024 * 2);
    __bf16* wf_f1  = (__bf16*)alloc((size_t)800 * 2048 * 2);
    __bf16* wf_f2  = (__bf16*)alloc((size_t)2048 * 256 * 2);
    (void)ws_size; (void)n_in; (void)out_size;

    const float NEG_INF = -__builtin_inff();
    auto fill = [&](float* p, float v, long long n) {
        k_fill<<<cdiv(n, 256), 256, 0, stream>>>(p, v, (int)n);
    };
    auto wfrag = [&](const float* W, __bf16* Wf, int Ksrc, int Kpad, int Nc) {
        k_w_frag<<<cdiv((long long)Kpad * Nc, 256), 256, 0, stream>>>(W, Wf, Ksrc, Kpad, Nc);
    };
    auto gemm = [&](const __bf16* A, const __bf16* Wf, const float* bias, void* D,
                    int M, int K, int Nc, int act, int obf) {
        long long waves = (long long)(M >> 4) * (Nc >> 6);
        k_gemm_wmma<<<cdiv(waves, 8), 256, 0, stream>>>(A, Wf, bias, D, M, K, Nc, act, obf);
    };

    // ---- weights -> bf16 fragment layout (padding ragged K) ----
    wfrag(pg1_w, wf_pg1, GF,   32,   256);
    wfrag(pg2_w, wf_pg2, 256,  256,  2048);
    wfrag(pg3_w, wf_pg3, 2048, 2048, 2048);
    wfrag(pg4_w, wf_pg4, 2048, 2048, 256);
    wfrag(g1_l_w, wf_g1l, 16,  32,   1024);
    wfrag(g1_r_w, wf_g1r, 16,  32,   1024);
    wfrag(g2_l_w, wf_g2l, 128, 128,  1024);
    wfrag(g2_r_w, wf_g2r, 128, 128,  1024);
    wfrag(f1_w,  wf_f1,  769,  800,  2048);
    wfrag(f2_w,  wf_f2,  2048, 2048, 256);
    k_pad_cols_bf16<<<cdiv((long long)Bg * 32, 256), 256, 0, stream>>>(y_in, ygpad, Bg, GF, 32);

    // ---- preprocessing ----
    fill(loopAttr, 0.f, N);
    k_loop_attr<<<cdiv(E, 256), 256, 0, stream>>>(ea_in, ei, loopAttr, E);
    k_pre_edge<<<cdiv((long long)EN * 2, 256), 256, 0, stream>>>(
        ea_in, loopAttr, pe1_w, pe1_b, bn_e_w, bn_e_b, ea2, E, EN);
    k_pre_node<<<cdiv((long long)N * 32, 256), 256, 0, stream>>>(
        x_in, pn1_w, pn1_b, bn_n_w, bn_n_b, xn, N);

    // ---- global MLP (WMMA) ----
    gemm(ygpad, wf_pg1, pg1_b, gbufA, Bg, 32,   256,  1, 1);
    gemm(gbufA, wf_pg2, pg2_b, gbufB, Bg, 256,  2048, 1, 1);
    gemm(gbufB, wf_pg3, pg3_b, gbufA, Bg, 2048, 2048, 1, 1);
    gemm(gbufA, wf_pg4, pg4_b, gfin,  Bg, 2048, 256,  1, 0);

    // ---- GATv2 layer 1 ----
    gemm(xn, wf_g1l, g1_l_b, xl, N, 32, 1024, 0, 1);
    gemm(xn, wf_g1r, g1_r_b, xr, N, 32, 1024, 0, 1);
    fill(smax, NEG_INF, (long long)N * 8);
    fill(ssum, 0.f, (long long)N * 8);
    fill(acc, 0.f, (long long)N * 1024);
    k_gat_score<<<EN, 256, 0, stream>>>(xl, xr, ei, E, EN, g1_att, nullptr, nullptr,
                                        alpha1, smax);
    k_softmax_exp<<<cdiv((long long)EN * 8, 256), 256, 0, stream>>>(alpha1, smax, ssum, ei, E, EN);
    k_softmax_norm<<<cdiv((long long)EN * 8, 256), 256, 0, stream>>>(alpha1, ssum, ei, E, EN);
    k_aggregate<<<EN, 256, 0, stream>>>(xl, alpha1, ei, E, EN, acc);
    k_head_mean<<<cdiv((long long)N * 128, 256), 256, 0, stream>>>(acc, g1_bias, h1, N);

    // SimpleConv(Mul) + leaky(0.01) -> bf16 A operand for layer-2 GEMMs
    fill(pbuf, 1.f, (long long)N * 128);
    k_scatter_mul<<<cdiv((long long)EN * 128, 256), 256, 0, stream>>>(h1, ei, E, EN, pbuf);
    k_leaky_to_bf16<<<cdiv((long long)N * 128, 256), 256, 0, stream>>>(pbuf, h2bf, N * 128, 0.01f);

    // ---- GATv2 layer 2 (edge_dim=8 via alpha1 @ g2_we) ----
    gemm(h2bf, wf_g2l, g2_l_b, xl, N, 128, 1024, 0, 1);
    gemm(h2bf, wf_g2r, g2_r_b, xr, N, 128, 1024, 0, 1);
    fill(smax, NEG_INF, (long long)N * 8);
    fill(ssum, 0.f, (long long)N * 8);
    fill(acc, 0.f, (long long)N * 1024);
    k_gat_score<<<EN, 256, 0, stream>>>(xl, xr, ei, E, EN, g2_att, g2_we, alpha1,
                                        alpha2, smax);
    k_softmax_exp<<<cdiv((long long)EN * 8, 256), 256, 0, stream>>>(alpha2, smax, ssum, ei, E, EN);
    k_softmax_norm<<<cdiv((long long)EN * 8, 256), 256, 0, stream>>>(alpha2, ssum, ei, E, EN);
    k_aggregate<<<EN, 256, 0, stream>>>(xl, alpha2, ei, E, EN, acc);
    k_head_mean<<<cdiv((long long)N * 128, 256), 256, 0, stream>>>(acc, g2_bias, ho, N);

    // SimpleConv([Sum,Mul]) -> concat(256) -> leaky(0.01)
    fill(msum, 0.f, (long long)N * 128);
    fill(mmul, 1.f, (long long)N * 128);
    k_scatter_sum_mul<<<cdiv((long long)EN * 128, 256), 256, 0, stream>>>(ho, ei, E, EN, msum, mmul);
    k_concat_leaky<<<cdiv((long long)N * 256, 256), 256, 0, stream>>>(msum, mmul, feat, N);

    // GraphNorm(256)
    k_graphnorm<<<Bg, 256, 0, stream>>>(feat, gn_w, gn_b, gn_ms, npg);

    // edge attention embedding -> et[EN]
    k_edge_embed<<<EN, 128, 0, stream>>>(ea2, alpha2, ee1_w, ee1_b, ee2_w, ee2_b, et, EN);

    // pooling
    k_node_pool<<<Bg, 256, 0, stream>>>(feat, pnn_w, pnn_b, pn, npg);
    k_csum<<<1, 1, 0, stream>>>(y_in, csum, Bg, GF, npg);
    k_edge_pool<<<Bg, 256, 0, stream>>>(et, csum, pen_w, pen_b, pe, EN);

    // final MLP (WMMA) + sigmoid
    k_build_h<<<cdiv((long long)Bg * 800, 256), 256, 0, stream>>>(pn, pe, gfin, hfeat, Bg);
    gemm(hfeat, wf_f1, f1_b, f1o, Bg, 800,  2048, 1, 1);
    gemm(f1o,   wf_f2, f2_b, f2o, Bg, 2048, 256,  1, 0);
    k_f3_sigmoid<<<cdiv(Bg, 64), 64, 0, stream>>>(f2o, f3_w, f3_b, (float*)d_out, Bg);
}
